// GCNModel_32203664785488
// MI455X (gfx1250) — matrix operand order repaired and checked by reading, not compiled, not run on hardware
//
#include <hip/hip_runtime.h>
#include <math.h>

#define N_NODES  100000
#define N_EDGES  300000
#define N_GRAPHS 2000
#define HD       256
#define C_IN     59
#define L_LAYERS 3
#define EPS_GN   1e-5f
#define LDSPAD   8          // 8 bf16 = 16B row padding -> rotates LDS bank groups

typedef __attribute__((ext_vector_type(16))) __bf16 v16bf;
typedef __attribute__((ext_vector_type(8)))  float  v8f;

// ---------------- small utility kernels ----------------

__global__ void count_kernel(const int* __restrict__ batch, float* __restrict__ counts, int n) {
    int i = blockIdx.x * blockDim.x + threadIdx.x;
    if (i < n) atomicAdd(&counts[batch[i]], 1.0f);
}

__global__ void deg_kernel(const int* __restrict__ dst, float* __restrict__ deg, int n) {
    int e = blockIdx.x * blockDim.x + threadIdx.x;
    if (e < n) atomicAdd(&deg[dst[e]], 1.0f);
}

__global__ void dinv_kernel(float* __restrict__ deg, int n) {
    int i = blockIdx.x * blockDim.x + threadIdx.x;
    if (i < n) deg[i] = rsqrtf(deg[i] + 1.0f);   // self-loop adds 1
}

// lin0: h = elu(x @ W0^T + b0), also emit bf16 copy for WMMA
__global__ __launch_bounds__(HD)
void lin0_kernel(const float* __restrict__ x, const float* __restrict__ W0,
                 const float* __restrict__ b0, float* __restrict__ h,
                 __bf16* __restrict__ hb) {
    __shared__ float sx[C_IN];
    int i = blockIdx.x, j = threadIdx.x;
    if (j < C_IN) sx[j] = x[(size_t)i * C_IN + j];
    __syncthreads();
    float acc = b0[j];
    const float* wrow = W0 + (size_t)j * C_IN;
    #pragma unroll 1
    for (int k = 0; k < C_IN; ++k) acc = fmaf(sx[k], wrow[k], acc);
    float v = acc > 0.0f ? acc : (expf(acc) - 1.0f);   // ELU
    size_t o = (size_t)i * HD + j;
    h[o]  = v;
    hb[o] = (__bf16)v;
}

// convert one layer's conv weights to bf16
__global__ void wconv_kernel(const float* __restrict__ W, __bf16* __restrict__ Wb) {
    int i = blockIdx.x * blockDim.x + threadIdx.x;   // HD*HD threads
    Wb[i] = (__bf16)W[i];
}

// ---------------- WMMA GEMM: t = hb @ Wb^T  ([N,256] x [256,256]^T) ----------------
// 128-thread block = 4 waves; block computes a full 16x256 output row-tile.
// A tile (16x256 bf16, contiguous 8KB) staged once in padded LDS; each wave
// computes 16x64 (4 accumulators), reusing the A fragment across 4 WMMAs/K-step.
__global__ __launch_bounds__(128)
void gemm_bf16_wmma(const __bf16* __restrict__ A,   // [N,256] row-major (h)
                    const __bf16* __restrict__ B,   // [256,256] row-major W (j,k)
                    float* __restrict__ C) {        // [N,256] f32
    __shared__ __bf16 sA[16 * (HD + LDSPAD)];       // row stride 264 bf16 = 33 uint4

    const int lane = threadIdx.x;            // 0..31
    const int wave = threadIdx.y;            // 0..3
    const int tid  = wave * 32 + lane;       // 0..127
    const int m0   = blockIdx.x * 16;        // node-tile

    // cooperative copy: global tile is contiguous (row stride == tile width)
    {
        const uint4* gsrc = (const uint4*)(A + (size_t)m0 * HD);  // 512 uint4
        uint4* sdst = (uint4*)sA;
        const int row = tid >> 3;            // 0..15
        const int seg = tid & 7;             // 0..7
        #pragma unroll
        for (int q = 0; q < 4; ++q) {
            int i = seg * 4 + q;             // 0..31 (32 uint4 per row)
            sdst[row * 33 + i] = gsrc[row * 32 + i];
        }
    }
    __syncthreads();

    const int ln   = lane & 15;
    const int half = lane >> 4;              // 0 or 1
    const __bf16* arow = sA + ln * (HD + LDSPAD);

    const int n0 = wave * 64;                // wave covers feature cols n0..n0+63
    const __bf16* brow0 = B + (size_t)(n0 +      ln) * HD + half * 16;
    const __bf16* brow1 = B + (size_t)(n0 + 16 + ln) * HD + half * 16;
    const __bf16* brow2 = B + (size_t)(n0 + 32 + ln) * HD + half * 16;
    const __bf16* brow3 = B + (size_t)(n0 + 48 + ln) * HD + half * 16;

    v8f acc0 = {}, acc1 = {}, acc2 = {}, acc3 = {};
    #pragma unroll
    for (int k = 0; k < HD; k += 32) {
        v16bf a;
        const __bf16* ap = arow + k + half * 16;     // contiguous K run of 16 per lane half (matches B)
        #pragma unroll
        for (int j = 0; j < 8; ++j) { a[j] = ap[j]; a[j + 8] = ap[8 + j]; }
        v16bf b0, b1, b2, b3;
        #pragma unroll
        for (int j = 0; j < 16; ++j) {
            b0[j] = brow0[k + j]; b1[j] = brow1[k + j];
            b2[j] = brow2[k + j]; b3[j] = brow3[k + j];
        }
        acc0 = __builtin_amdgcn_wmma_f32_16x16x32_bf16(false, a, false, b0, (short)0, acc0, false, false);
        acc1 = __builtin_amdgcn_wmma_f32_16x16x32_bf16(false, a, false, b1, (short)0, acc1, false, false);
        acc2 = __builtin_amdgcn_wmma_f32_16x16x32_bf16(false, a, false, b2, (short)0, acc2, false, false);
        acc3 = __builtin_amdgcn_wmma_f32_16x16x32_bf16(false, a, false, b3, (short)0, acc3, false, false);
    }
    // C/D layout: lanes 0-15 -> M = r, lanes 16-31 -> M = r+8; N = lane&15
    float* crow = C + (size_t)(m0 + half * 8) * HD + ln;
    #pragma unroll
    for (int r = 0; r < 8; ++r) {
        crow[(size_t)r * HD + n0     ] = acc0[r];
        crow[(size_t)r * HD + n0 + 16] = acc1[r];
        crow[(size_t)r * HD + n0 + 32] = acc2[r];
        crow[(size_t)r * HD + n0 + 48] = acc3[r];
    }
}

// ---------------- GCN aggregation ----------------

// agg[dst] += coef * t[src]  (agg aliases h buffer, pre-zeroed)
__global__ __launch_bounds__(HD)
void scatter_kernel(const int* __restrict__ src, const int* __restrict__ dst,
                    const float* __restrict__ dinv, const float* __restrict__ t,
                    float* __restrict__ agg) {
    int e = blockIdx.x, j = threadIdx.x;
    int s = src[e], d = dst[e];
    float coef = dinv[s] * dinv[d];
    atomicAdd(&agg[(size_t)d * HD + j], t[(size_t)s * HD + j] * coef);
}

// h = agg + t*dinv^2 + b  ; accumulate graph mean numerator
__global__ __launch_bounds__(HD)
void selfbias_mean_kernel(float* __restrict__ h, const float* __restrict__ t,
                          const float* __restrict__ dinv, const float* __restrict__ b,
                          const int* __restrict__ batch, float* __restrict__ mean) {
    int i = blockIdx.x, j = threadIdx.x;
    size_t o = (size_t)i * HD + j;
    float di = dinv[i];
    float v = h[o] + t[o] * di * di + b[j];
    h[o] = v;
    atomicAdd(&mean[(size_t)batch[i] * HD + j], v);
}

__global__ void mean_div_kernel(float* __restrict__ mean, const float* __restrict__ counts) {
    int idx = blockIdx.x * blockDim.x + threadIdx.x;     // G*HD threads
    float c = counts[idx >> 8]; c = c < 1.0f ? 1.0f : c;
    mean[idx] /= c;
}

// out = h - alpha*mean[batch]; accumulate variance numerator
__global__ __launch_bounds__(HD)
void center_var_kernel(float* __restrict__ h, const float* __restrict__ mean,
                       const float* __restrict__ alpha, const int* __restrict__ batch,
                       float* __restrict__ var) {
    int i = blockIdx.x, j = threadIdx.x;
    size_t o = (size_t)i * HD + j;
    int bg = batch[i];
    float v = h[o] - alpha[j] * mean[(size_t)bg * HD + j];
    h[o] = v;
    atomicAdd(&var[(size_t)bg * HD + j], v * v);
}

__global__ void rstd_kernel(float* __restrict__ var, const float* __restrict__ counts) {
    int idx = blockIdx.x * blockDim.x + threadIdx.x;     // G*HD threads
    float c = counts[idx >> 8]; c = c < 1.0f ? 1.0f : c;
    var[idx] = rsqrtf(var[idx] / c + EPS_GN);
}

// h = relu(h*rstd*gamma + beta); refresh bf16 copy for next GEMM
__global__ __launch_bounds__(HD)
void norm_relu_kernel(float* __restrict__ h, __bf16* __restrict__ hb,
                      const float* __restrict__ rstd, const float* __restrict__ gamma,
                      const float* __restrict__ beta, const int* __restrict__ batch) {
    int i = blockIdx.x, j = threadIdx.x;
    size_t o = (size_t)i * HD + j;
    float v = h[o] * rstd[(size_t)batch[i] * HD + j] * gamma[j] + beta[j];
    v = v > 0.0f ? v : 0.0f;
    h[o]  = v;
    hb[o] = (__bf16)v;
}

// global_add_pool
__global__ __launch_bounds__(HD)
void pool_kernel(const float* __restrict__ h, const int* __restrict__ batch,
                 float* __restrict__ g) {
    int i = blockIdx.x, j = threadIdx.x;
    atomicAdd(&g[(size_t)batch[i] * HD + j], h[(size_t)i * HD + j]);
}

// head: sigmoid(relu(g@W1^T + b1) @ Wo^T + bo)
__global__ __launch_bounds__(HD)
void head_kernel(const float* __restrict__ g, const float* __restrict__ W1,
                 const float* __restrict__ b1, const float* __restrict__ Wo,
                 const float* __restrict__ bo, float* __restrict__ out) {
    __shared__ float sg[HD];
    __shared__ float red[HD];
    int gr = blockIdx.x, j = threadIdx.x;
    sg[j] = g[(size_t)gr * HD + j];
    __syncthreads();
    float acc = b1[j];
    const float* wrow = W1 + (size_t)j * HD;
    #pragma unroll 4
    for (int k = 0; k < HD; ++k) acc = fmaf(sg[k], wrow[k], acc);
    float r = acc > 0.0f ? acc : 0.0f;
    red[j] = r * Wo[j];
    __syncthreads();
    for (int s = HD / 2; s > 0; s >>= 1) {
        if (j < s) red[j] += red[j + s];
        __syncthreads();
    }
    if (j == 0) out[gr] = 1.0f / (1.0f + expf(-(red[0] + bo[0])));
}

// ---------------- launch ----------------

extern "C" void kernel_launch(void* const* d_in, const int* in_sizes, int n_in,
                              void* d_out, int out_size, void* d_ws, size_t ws_size,
                              hipStream_t stream) {
    (void)in_sizes; (void)n_in; (void)out_size; (void)ws_size;

    const float* x       = (const float*)d_in[0];
    const int*   ei      = (const int*)  d_in[1];
    const int*   batch   = (const int*)  d_in[2];
    const float* lin0_W  = (const float*)d_in[3];
    const float* lin0_b  = (const float*)d_in[4];
    const float* conv_W  = (const float*)d_in[5];
    const float* conv_b  = (const float*)d_in[6];
    const float* alpha   = (const float*)d_in[7];
    const float* gamma   = (const float*)d_in[8];
    const float* beta    = (const float*)d_in[9];
    const float* lin1_W  = (const float*)d_in[10];
    const float* lin1_b  = (const float*)d_in[11];
    const float* out_W   = (const float*)d_in[12];
    const float* out_b   = (const float*)d_in[13];
    float* out = (float*)d_out;

    const int* src = ei;
    const int* dst = ei + N_EDGES;

    // workspace carve-up
    char* ws = (char*)d_ws;
    size_t ofs = 0;
    float*  h     = (float*)(ws + ofs); ofs += (size_t)N_NODES * HD * 4;   // also agg
    float*  t     = (float*)(ws + ofs); ofs += (size_t)N_NODES * HD * 4;
    __bf16* hb    = (__bf16*)(ws + ofs); ofs += (size_t)N_NODES * HD * 2;
    __bf16* Wb    = (__bf16*)(ws + ofs); ofs += (size_t)HD * HD * 2;
    float*  dinv  = (float*)(ws + ofs); ofs += (size_t)N_NODES * 4;        // deg then dinv
    float*  cnts  = (float*)(ws + ofs); ofs += (size_t)N_GRAPHS * 4;
    float*  mean  = (float*)(ws + ofs); ofs += (size_t)N_GRAPHS * HD * 4;
    float*  rstd  = (float*)(ws + ofs); ofs += (size_t)N_GRAPHS * HD * 4;  // var then rstd
    float*  gpool = (float*)(ws + ofs); ofs += (size_t)N_GRAPHS * HD * 4;

    const int T = 256;
    hipMemsetAsync(cnts, 0, (size_t)N_GRAPHS * 4, stream);
    hipMemsetAsync(dinv, 0, (size_t)N_NODES * 4, stream);
    count_kernel<<<(N_NODES + T - 1) / T, T, 0, stream>>>(batch, cnts, N_NODES);
    deg_kernel<<<(N_EDGES + T - 1) / T, T, 0, stream>>>(dst, dinv, N_EDGES);
    dinv_kernel<<<(N_NODES + T - 1) / T, T, 0, stream>>>(dinv, N_NODES);

    lin0_kernel<<<N_NODES, HD, 0, stream>>>(x, lin0_W, lin0_b, h, hb);

    for (int l = 0; l < L_LAYERS; ++l) {
        wconv_kernel<<<(HD * HD) / T, T, 0, stream>>>(conv_W + (size_t)l * HD * HD, Wb);
        gemm_bf16_wmma<<<N_NODES / 16, dim3(32, 4), 0, stream>>>(hb, Wb, t);
        hipMemsetAsync(h, 0, (size_t)N_NODES * HD * 4, stream);
        scatter_kernel<<<N_EDGES, HD, 0, stream>>>(src, dst, dinv, t, h);
        hipMemsetAsync(mean, 0, (size_t)N_GRAPHS * HD * 4, stream);
        selfbias_mean_kernel<<<N_NODES, HD, 0, stream>>>(h, t, dinv, conv_b + (size_t)l * HD,
                                                         batch, mean);
        mean_div_kernel<<<(N_GRAPHS * HD) / T, T, 0, stream>>>(mean, cnts);
        hipMemsetAsync(rstd, 0, (size_t)N_GRAPHS * HD * 4, stream);
        center_var_kernel<<<N_NODES, HD, 0, stream>>>(h, mean, alpha + (size_t)l * HD,
                                                      batch, rstd);
        rstd_kernel<<<(N_GRAPHS * HD) / T, T, 0, stream>>>(rstd, cnts);
        norm_relu_kernel<<<N_NODES, HD, 0, stream>>>(h, hb, rstd, gamma + (size_t)l * HD,
                                                     beta + (size_t)l * HD, batch);
    }

    hipMemsetAsync(gpool, 0, (size_t)N_GRAPHS * HD * 4, stream);
    pool_kernel<<<N_NODES, HD, 0, stream>>>(h, batch, gpool);
    head_kernel<<<N_GRAPHS, HD, 0, stream>>>(gpool, lin1_W, lin1_b, out_W, out_b, out);
}
